// NCF_80212809220384
// MI455X (gfx1250) — matrix operand order repaired and checked by reading, not compiled, not run on hardware
//
#include <hip/hip_runtime.h>
#include <hip/hip_bf16.h>
#include <math.h>

// ---------------------------------------------------------------------------
// NCF fused forward for MI455X (gfx1250, wave32, WMMA f32_16x16x32_f16)
//   eu/ei gathers -> GMF product folded into row partial sums
//   concat(mlp) @ W1 -> relu -> @ W2 -> relu -> @ W3 -> relu
//   sigmoid(gmf.Wp + h.Wp + bp)
// One block = 64 batch rows, 8 waves; each wave owns one 16-col output tile
// per GEMM and keeps the corresponding weight B-fragments in registers.
// ---------------------------------------------------------------------------

typedef __attribute__((ext_vector_type(16))) _Float16 v16h;
typedef __attribute__((ext_vector_type(8)))  _Float16 v8h;
typedef __attribute__((ext_vector_type(8)))  float    v8f;

#define BM   64      // batch rows per block
#define XP   136     // LDS pitch for 128-wide f16 activations (padded)
#define H1P  136
#define H2P  72      // pitch for 64-wide f16
#define H3P  33      // pitch for 32-wide f32

__device__ __forceinline__ float relu_(float x) { return fmaxf(x, 0.f); }

// A-fragment (16x32 f16) from an LDS row, contiguous K order:
//  lanes 0-15 : K in {kb..kb+15}
//  lanes 16-31: K in {kb+16..kb+31}
__device__ __forceinline__ v16h load_a_frag(const _Float16* row, int kbase, int khalf) {
    union { v16h v; v8h h[2]; } u;
    u.h[0] = *(const v8h*)(row + kbase + khalf * 16);
    u.h[1] = *(const v8h*)(row + kbase + khalf * 16 + 8);
    return u.v;
}

// B-fragment (32x16 f16) built in registers from row-major f32 W[K][N] in
// global (L2-hot): lane holds column `col`, elements e -> K = kbase+khalf*16+e.
__device__ __forceinline__ v16h load_b_frag(const float* __restrict__ W, int N,
                                            int col, int kbase, int khalf) {
    v16h b;
    const float* p = W + (size_t)(kbase + khalf * 16) * N + col;
#pragma unroll
    for (int e = 0; e < 16; ++e) b[e] = (_Float16)p[(size_t)e * N];
    return b;
}

extern "C" __global__ __launch_bounds__(256)
void ncf_fused(const int* __restrict__ user, const int* __restrict__ item,
               const float* __restrict__ Wug, const float* __restrict__ bug,
               const float* __restrict__ Wum, const float* __restrict__ bum,
               const float* __restrict__ Wig, const float* __restrict__ big,
               const float* __restrict__ Wim, const float* __restrict__ bim,
               const float* __restrict__ W1,  const float* __restrict__ b1,
               const float* __restrict__ W2,  const float* __restrict__ b2,
               const float* __restrict__ W3,  const float* __restrict__ b3,
               const float* __restrict__ Wp,  const float* __restrict__ bp,
               float* __restrict__ out, int batch) {
    __shared__ _Float16 Xs [BM * XP];   // relu(concat(eu_mlp, ei_mlp)) f16
    __shared__ _Float16 H1s[BM * H1P];  // relu(X @ W1 + b1)            f16
    __shared__ _Float16 H2s[BM * H2P];  // relu(H1 @ W2 + b2)           f16
    __shared__ float    H3s[BM * H3P];  // relu(H2 @ W3 + b3)           f32
    __shared__ float    Gp [BM * 4];    // GMF.Wp partial sums (4 per row)

    const int t     = threadIdx.x;
    const int lane  = t & 31;
    const int w     = t >> 5;        // wave id 0..7
    const int laneN = lane & 15;     // output column within tile / A row
    const int khalf = lane >> 4;     // which K-half this lane holds
    const int blk   = blockIdx.x;

    // ---- per-wave weight B-fragments in registers (tiny, L2-resident) ----
    const int col1 = w * 16 + laneN;          // GEMM1: col tile = wave id
    const int col2 = (w & 3) * 16 + laneN;    // GEMM2: 4 col tiles
    const int col3 = (w & 1) * 16 + laneN;    // GEMM3: 2 col tiles
    v16h B1[4], B2[4], B3[2];
#pragma unroll
    for (int kb = 0; kb < 4; ++kb) B1[kb] = load_b_frag(W1, 128, col1, kb * 32, khalf);
#pragma unroll
    for (int kb = 0; kb < 4; ++kb) B2[kb] = load_b_frag(W2, 64, col2, kb * 32, khalf);
#pragma unroll
    for (int kb = 0; kb < 2; ++kb) B3[kb] = load_b_frag(W3, 32, col3, kb * 32, khalf);
    const float bias1 = b1[col1];
    const float bias2 = b2[col2];
    const float bias3 = b3[col3];

    // ---- gather embeddings: 4 threads per row, 16 cols each ----
    {
        const int r = t >> 2;                     // 0..63
        const int q = t & 3;                      // 16-col chunk
        int g = blk * BM + r;
        if (g >= batch) g = batch - 1;            // clamp (grid covers batch)
        const int u  = user[g];
        const int it = item[g];
        const float4* ug4  = (const float4*)(Wug + (size_t)u  * 64 + q * 16);
        const float4* um4  = (const float4*)(Wum + (size_t)u  * 64 + q * 16);
        const float4* ig4  = (const float4*)(Wig + (size_t)it * 64 + q * 16);
        const float4* im4  = (const float4*)(Wim + (size_t)it * 64 + q * 16);
        const float4* bug4 = (const float4*)(bug + q * 16);
        const float4* bum4 = (const float4*)(bum + q * 16);
        const float4* big4 = (const float4*)(big + q * 16);
        const float4* bim4 = (const float4*)(bim + q * 16);
        const float4* wp4  = (const float4*)(Wp  + q * 16);

        union F4 { float4 v; float f[4]; };
        float gsum = 0.f;
#pragma unroll
        for (int i = 0; i < 4; ++i) {
            F4 aug{ug4[i]}, aig{ig4[i]}, aum{um4[i]}, aim{im4[i]};
            F4 cbug{bug4[i]}, cbig{big4[i]}, cbum{bum4[i]}, cbim{bim4[i]};
            F4 cwp{wp4[i]};
#pragma unroll
            for (int j = 0; j < 4; ++j) {
                const int c = q * 16 + i * 4 + j;
                float egu = relu_(aug.f[j] + cbug.f[j]);
                float egi = relu_(aig.f[j] + cbig.f[j]);
                gsum += egu * egi * cwp.f[j];     // GMF path folded into Wp dot
                float emu = relu_(aum.f[j] + cbum.f[j]);
                float emi = relu_(aim.f[j] + cbim.f[j]);
                Xs[r * XP + c]      = (_Float16)emu;   // cols 0..63  = eu_mlp
                Xs[r * XP + 64 + c] = (_Float16)emi;   // cols 64..127= ei_mlp
            }
        }
        Gp[r * 4 + q] = gsum;   // deterministic partial (no atomics)
    }
    __syncthreads();

    // ---- GEMM1: H1[64x128] = relu(X[64x128] @ W1[128x128] + b1) ----
#pragma unroll
    for (int rt = 0; rt < 4; ++rt) {
        v8f acc;
#pragma unroll
        for (int g = 0; g < 8; ++g) acc[g] = bias1;
        const _Float16* arow = &Xs[(rt * 16 + laneN) * XP];
#pragma unroll
        for (int kb = 0; kb < 4; ++kb) {
            v16h a = load_a_frag(arow, kb * 32, khalf);
            acc = __builtin_amdgcn_wmma_f32_16x16x32_f16(
                false, a, false, B1[kb], (short)0, acc, false, false);
        }
#pragma unroll
        for (int g = 0; g < 8; ++g)
            H1s[(rt * 16 + g + khalf * 8) * H1P + col1] = (_Float16)relu_(acc[g]);
    }
    __syncthreads();

    // ---- GEMM2: H2[64x64] = relu(H1 @ W2[128x64] + b2) ----
    {
        const int rt0 = (w >> 2) * 2;
#pragma unroll
        for (int rr = 0; rr < 2; ++rr) {
            const int rt = rt0 + rr;
            v8f acc;
#pragma unroll
            for (int g = 0; g < 8; ++g) acc[g] = bias2;
            const _Float16* arow = &H1s[(rt * 16 + laneN) * H1P];
#pragma unroll
            for (int kb = 0; kb < 4; ++kb) {
                v16h a = load_a_frag(arow, kb * 32, khalf);
                acc = __builtin_amdgcn_wmma_f32_16x16x32_f16(
                    false, a, false, B2[kb], (short)0, acc, false, false);
            }
#pragma unroll
            for (int g = 0; g < 8; ++g)
                H2s[(rt * 16 + g + khalf * 8) * H2P + col2] = (_Float16)relu_(acc[g]);
        }
    }
    __syncthreads();

    // ---- GEMM3: H3[64x32] = relu(H2 @ W3[64x32] + b3) ----
    {
        const int rt = w >> 1;
        v8f acc;
#pragma unroll
        for (int g = 0; g < 8; ++g) acc[g] = bias3;
        const _Float16* arow = &H2s[(rt * 16 + laneN) * H2P];
#pragma unroll
        for (int kb = 0; kb < 2; ++kb) {
            v16h a = load_a_frag(arow, kb * 32, khalf);
            acc = __builtin_amdgcn_wmma_f32_16x16x32_f16(
                false, a, false, B3[kb], (short)0, acc, false, false);
        }
#pragma unroll
        for (int g = 0; g < 8; ++g)
            H3s[(rt * 16 + g + khalf * 8) * H3P + col3] = relu_(acc[g]);
    }
    __syncthreads();

    // ---- final: sigmoid(gmf.Wp[0:64] + h.Wp[64:96] + bp) ----
    if (t < BM) {
        const int g = blk * BM + t;
        if (g < batch) {
            float s = bp[0] + Gp[t * 4 + 0] + Gp[t * 4 + 1]
                            + Gp[t * 4 + 2] + Gp[t * 4 + 3];
#pragma unroll
            for (int k = 0; k < 32; ++k) s += H3s[t * H3P + k] * Wp[64 + k];
            out[g] = 1.f / (1.f + expf(-s));
        }
    }
}

extern "C" void kernel_launch(void* const* d_in, const int* in_sizes, int n_in,
                              void* d_out, int out_size, void* d_ws, size_t ws_size,
                              hipStream_t stream) {
    const int*   user = (const int*)d_in[0];
    const int*   item = (const int*)d_in[1];
    const float* Wug  = (const float*)d_in[2];
    const float* bug  = (const float*)d_in[3];
    const float* Wum  = (const float*)d_in[4];
    const float* bum  = (const float*)d_in[5];
    const float* Wig  = (const float*)d_in[6];
    const float* big  = (const float*)d_in[7];
    const float* Wim  = (const float*)d_in[8];
    const float* bim  = (const float*)d_in[9];
    const float* W1   = (const float*)d_in[10];
    const float* b1   = (const float*)d_in[11];
    const float* W2   = (const float*)d_in[12];
    const float* b2   = (const float*)d_in[13];
    const float* W3   = (const float*)d_in[14];
    const float* b3   = (const float*)d_in[15];
    const float* Wp   = (const float*)d_in[16];
    const float* bp   = (const float*)d_in[17];
    float* out = (float*)d_out;

    const int batch = in_sizes[0];
    const int grid  = (batch + BM - 1) / BM;
    ncf_fused<<<grid, 256, 0, stream>>>(user, item, Wug, bug, Wum, bum,
                                        Wig, big, Wim, bim,
                                        W1, b1, W2, b2, W3, b3, Wp, bp,
                                        out, batch);
}
